// BinarizeConv2dSDP_592705486860
// MI455X (gfx1250) — compile-verified, matrix operand order checked by reading, not run on hardware
//
#include <hip/hip_runtime.h>

// Binarized conv2d (XNOR-net forward) for MI455X / gfx1250.
//   out = conv2d(sign(x), sign(M), stride=1, pad=1) * Alpha
// Implicit GEMM on V_WMMA_I32_16X16X64_IU8 (exact integer math == f32 ref).
// Block tile: 128 pixels (8 waves x 16) x 128 out-channels (8 WMMA tiles per
// wave). B panel K-chunk staged into LDS (48 KB x 6 chunks) via
// GLOBAL_LOAD_ASYNC_TO_LDS_B128 (ASYNCcnt-drained) when available.

typedef __attribute__((ext_vector_type(8))) int v8i;
typedef int v4i_raw __attribute__((vector_size(16)));  // native clang vector

#if __has_builtin(__builtin_amdgcn_global_load_async_to_lds_b128)
#define USE_ASYNC_LDS 1
typedef __attribute__((address_space(1))) v4i_raw gas_v4i;  // global
typedef __attribute__((address_space(3))) v4i_raw las_v4i;  // LDS
#else
#define USE_ASYNC_LDS 0
#endif

#define AN 32
#define AC 256
#define AH 56
#define AW 56
#define OC 256
#define HW (AH * AW)          // 3136
#define NPIX (AN * HW)        // 100352 output pixels
#define NPTILE (NPIX / 16)    // 6272 pixel tiles (exact)
#define NKT 36                // K = 256*9 = 2304 = 36 steps of 64
#define NOT 16                // 16 channel tiles of 16
#define OLPB 8                // channel tiles per block (Nb = 128)
#define KCH 6                 // kt per LDS chunk (6 chunks of 48 KB)

#define ACT_BYTES ((size_t)NPIX * AC)              // 25,690,112
#define WPACK_BYTES ((size_t)NOT * NKT * 32 * 32)  // 589,824

// ---------------------------------------------------------------------------
// Pass 1: x (NCHW f32) -> act (NHWC int8 sign). LDS-tiled transpose:
// coalesced f32 reads along W, char4 writes along C.
// ---------------------------------------------------------------------------
__global__ __launch_bounds__(256) void k_bin_acts(const float* __restrict__ x,
                                                  signed char* __restrict__ act) {
  __shared__ signed char tile[32][33];
  const int bw = blockIdx.x & 1;
  const int bc = blockIdx.x >> 1;
  const int nh = blockIdx.y;
  const int n = nh / AH, h = nh % AH;
  const int tx = threadIdx.x & 31;
  const int ty = threadIdx.x >> 5;
  const int w0 = bw * 32, c0 = bc * 32;
#pragma unroll
  for (int i = 0; i < 4; ++i) {
    const int c = ty + 8 * i;
    const int w = w0 + tx;
    signed char s = 0;
    if (w < AW) {
      const float v = x[((n * AC + c0 + c) * AH + h) * AW + w];
      s = (v > 0.0f) ? 1 : ((v < 0.0f) ? -1 : 0);
    }
    tile[c][tx] = s;
  }
  __syncthreads();
  const int wl = threadIdx.x >> 3;
  const int cq = threadIdx.x & 7;
  const int w = w0 + wl;
  if (w < AW) {
    char4 p;
    p.x = tile[cq * 4 + 0][wl];
    p.y = tile[cq * 4 + 1][wl];
    p.z = tile[cq * 4 + 2][wl];
    p.w = tile[cq * 4 + 3][wl];
    *reinterpret_cast<char4*>(&act[((size_t)(n * AH + h) * AW + w) * AC + c0 + cq * 4]) = p;
  }
}

// ---------------------------------------------------------------------------
// Pass 2: M (OIHW f32) -> wp: per-lane WMMA IU8 B-fragments, pre-swizzled.
// K = (kh*3+kw)*256 + ic. Fragment layout mirrors the ISA 8-bit A layout:
//   lane<16:  v0..v7 hold K+{0-3,4-7,16-19,20-23,32-35,36-39,48-51,52-55}
//   lane>=16: same +8.
// wp[((ot*36+kt)*32+lane)*32 .. +31] -> 2x b128 per lane.
// ---------------------------------------------------------------------------
__global__ __launch_bounds__(256) void k_bin_wts(const float* __restrict__ M,
                                                 signed char* __restrict__ wp) {
  const int gid = blockIdx.x * 256 + threadIdx.x;
  if (gid >= NOT * NKT * 32) return;
  const int lane = gid & 31;
  const int kt = (gid >> 5) % NKT;
  const int ot = gid / (NKT * 32);
  const int half = lane >> 4;
  const int o = ot * 16 + (lane & 15);
  int frag[8];
#pragma unroll
  for (int v = 0; v < 8; ++v) {
    const int koff = (v >> 1) * 16 + half * 8 + (v & 1) * 4;
    unsigned int dw = 0;
#pragma unroll
    for (int j = 0; j < 4; ++j) {
      const int k = kt * 64 + koff + j;
      const int khkw = k >> 8;
      const int ic = k & 255;
      const int kh = khkw / 3, kw = khkw % 3;
      const float val = M[((o * AC + ic) * 3 + kh) * 3 + kw];
      const int s = (val > 0.0f) ? 1 : ((val < 0.0f) ? -1 : 0);
      dw |= ((unsigned int)(unsigned char)(signed char)s) << (8 * j);
    }
    frag[v] = (int)dw;
  }
  int4* dst = reinterpret_cast<int4*>(wp + ((size_t)(ot * NKT + kt) * 32 + lane) * 32);
  dst[0] = make_int4(frag[0], frag[1], frag[2], frag[3]);
  dst[1] = make_int4(frag[4], frag[5], frag[6], frag[7]);
}

// ---------------------------------------------------------------------------
// Pass 3: implicit-GEMM binarized conv on WMMA IU8.
// blockIdx -> (otg: 128-channel group, 8 pixel tiles). Each wave: 16 pixels x
// 128 channels = 8 accumulator tiles, 36 kt x 8 = 288 WMMA. B staged in LDS
// in 6 chunks of (8 ot x 6 kt) = 48 KB, async-to-LDS when available.
// Epilogue transposes each 16x16 tile through LDS -> contiguous float4 NCHW.
// ---------------------------------------------------------------------------
__global__ __launch_bounds__(256) void k_bconv_wmma(const signed char* __restrict__ act,
                                                    const signed char* __restrict__ wp,
                                                    const float* __restrict__ alpha,
                                                    float* __restrict__ out) {
  __shared__ __align__(32) int lds_b[OLPB * KCH * 32 * 8];  // 49,152 B
  __shared__ float ctile[8][16][17];                        //  8,704 B

  const int otg = blockIdx.x & 1;          // channel group: 0 or 1 (128 ch)
  const int ptg = blockIdx.x >> 1;         // 0..783
  const int tid = threadIdx.x;
  const int lane = tid & 31;
  const int wave = tid >> 5;
  const int ptile = ptg * 8 + wave;        // 0..6271
  const int row = lane & 15;
  const int halfofs = (lane >> 4) * 8;

  // This lane's A-row pixel (linear 16-pixel tiles; 16 | 3136, no n-crossing).
  const int pm = ptile * 16 + row;
  const int n = pm / HW;
  const int rem = pm - n * HW;
  const int oh = rem / AW;
  const int ow = rem - oh * AW;

  const v4i_raw* wp4 = reinterpret_cast<const v4i_raw*>(wp);
  const size_t wbase = (size_t)otg * OLPB * (NKT * 64);  // 16B units

  v8i acc[OLPB] = {};

  for (int c = 0; c < NKT / KCH; ++c) {  // 6 chunks
    if (c) __syncthreads();              // previous chunk's LDS reads done

    // ---- stage (8 ot x 6 kt) B chunk: 3072 x 16B, 12 per thread ----
#pragma unroll
    for (int j = 0; j < 12; ++j) {
      const int i = tid + 256 * j;       // dst 16B index in lds_b (linear)
      const int seg = i >> 6;            // (ol*KCH + kl)
      const int wi = i & 63;
      const int ol = seg / KCH;
      const int kl = seg - ol * KCH;
      const v4i_raw* src = wp4 + wbase + (size_t)ol * (NKT * 64) +
                           (size_t)(c * KCH + kl) * 64 + wi;
#if USE_ASYNC_LDS
      __builtin_amdgcn_global_load_async_to_lds_b128(
          (gas_v4i*)src, (las_v4i*)(reinterpret_cast<v4i_raw*>(lds_b) + i), 0, 0);
#else
      reinterpret_cast<v4i_raw*>(lds_b)[i] = *src;
#endif
    }
#if USE_ASYNC_LDS
#if __has_builtin(__builtin_amdgcn_s_wait_asynccnt)
    __builtin_amdgcn_s_wait_asynccnt(0);
#else
    asm volatile("s_wait_asynccnt 0x0" ::: "memory");
#endif
#endif
    __syncthreads();

    // ---- compute: 6 kt, 8 WMMA each (one A fragment reused 8x) ----
#pragma unroll
    for (int kl = 0; kl < KCH; ++kl) {
      const int kt = c * KCH + kl;
      const int tap = kt >> 2;           // 64 | 256: never crosses a tap
      const int kh = tap / 3, kw = tap - kh * 3;
      const int ic0 = (kt & 3) * 64;
      const int ih = oh + kh - 1;
      const int iw = ow + kw - 1;
      v8i a = {};
      if ((unsigned)ih < (unsigned)AH && (unsigned)iw < (unsigned)AW) {
        const uint2* ap = reinterpret_cast<const uint2*>(
            act + (((size_t)(n * AH + ih) * AW + iw) << 8) + ic0 + halfofs);
        const uint2 g0 = ap[0], g1 = ap[2], g2 = ap[4], g3 = ap[6];
        a[0] = (int)g0.x; a[1] = (int)g0.y;
        a[2] = (int)g1.x; a[3] = (int)g1.y;
        a[4] = (int)g2.x; a[5] = (int)g2.y;
        a[6] = (int)g3.x; a[7] = (int)g3.y;
      }  // else zero rows == conv zero-padding, exact

      // 8 channel tiles; reuse_a hint on the first 7 identical consecutive
      // WMMAs sharing the A operand (literal immediates as required).
#define WMMA_OL(OL, RA)                                                        \
  {                                                                            \
    const v8i b = *reinterpret_cast<const v8i*>(                               \
        &lds_b[(((OL) * KCH + kl) * 32 + lane) * 8]);                          \
    acc[(OL)] = __builtin_amdgcn_wmma_i32_16x16x64_iu8(true, a, true, b,       \
                                                       acc[(OL)], RA, false);  \
  }
      WMMA_OL(0, true)
      WMMA_OL(1, true)
      WMMA_OL(2, true)
      WMMA_OL(3, true)
      WMMA_OL(4, true)
      WMMA_OL(5, true)
      WMMA_OL(6, true)
      WMMA_OL(7, false)
#undef WMMA_OL
    }
  }

  // ---- epilogue: per channel tile, alpha-scale + LDS transpose ----
  const int pm0 = ptile * 16;
  const int n2 = pm0 / HW;
  const int hw0 = (pm0 - n2 * HW) + halfofs;  // multiple of 8 -> 32B aligned
#pragma unroll
  for (int ol = 0; ol < OLPB; ++ol) {
    const int chbase = otg * 128 + ol * 16;
    const float al = alpha[chbase + row];
#pragma unroll
    for (int r = 0; r < 8; ++r)
      ctile[wave][row][r + halfofs] = al * (float)acc[ol][r];  // [chan][pixel]
    __syncthreads();
    float* op = out + (size_t)(n2 * OC + chbase + row) * HW + hw0;
    float4 s0, s1;
    s0.x = ctile[wave][row][halfofs + 0];
    s0.y = ctile[wave][row][halfofs + 1];
    s0.z = ctile[wave][row][halfofs + 2];
    s0.w = ctile[wave][row][halfofs + 3];
    s1.x = ctile[wave][row][halfofs + 4];
    s1.y = ctile[wave][row][halfofs + 5];
    s1.z = ctile[wave][row][halfofs + 6];
    s1.w = ctile[wave][row][halfofs + 7];
    *reinterpret_cast<float4*>(op) = s0;
    *reinterpret_cast<float4*>(op + 4) = s1;
    __syncthreads();  // ctile reused by next channel tile
  }
}

// ---------------------------------------------------------------------------
extern "C" void kernel_launch(void* const* d_in, const int* in_sizes, int n_in,
                              void* d_out, int out_size, void* d_ws, size_t ws_size,
                              hipStream_t stream) {
  const float* x = (const float*)d_in[0];      // (32,256,56,56)
  const float* M = (const float*)d_in[1];      // (256,256,3,3)
  const float* alpha = (const float*)d_in[2];  // (256)
  float* out = (float*)d_out;                  // (32,256,56,56)

  signed char* act = (signed char*)d_ws;                // 25,690,112 B
  signed char* wpk = (signed char*)d_ws + ACT_BYTES;    // 589,824 B

  k_bin_acts<<<dim3(16, AN * AH), 256, 0, stream>>>(x, act);
  k_bin_wts<<<dim3((NOT * NKT * 32 + 255) / 256), 256, 0, stream>>>(M, wpk);
  // 784 pixel-tile groups x 2 channel groups
  k_bconv_wmma<<<dim3((NPTILE / 8) * 2), 256, 0, stream>>>(act, wpk, alpha, out);
}
